// MultiHeadAttention_23192823399120
// MI455X (gfx1250) — matrix operand order repaired and checked by reading, not compiled, not run on hardware
//
#include <hip/hip_runtime.h>
#include <hip/hip_bf16.h>

// ---------------------------------------------------------------------------
// MultiHeadAttention for MI455X (gfx1250): bf16 WMMA, f32 accumulate.
//   B=4, S=2048, D=1024, HEADS=16, DK=64. scale 1/8 folded into Q projection.
//   V is stored transposed ([b,h,d,S]) by the projection so attention needs
//   no in-kernel transpose. K/V(T) tiles staged to LDS via the Tensor Data
//   Mover when available.
// ---------------------------------------------------------------------------

typedef __bf16 bf16;
typedef __attribute__((ext_vector_type(16))) __bf16 v16bf;
typedef __attribute__((ext_vector_type(8)))  float  v8f;

union FragBF {
    v16bf v;
    uint2 u2[4];
};

__device__ __forceinline__ v8f wmma_bf16(v16bf a, v16bf b, v8f c) {
    return __builtin_amdgcn_wmma_f32_16x16x32_bf16(
        false, a, false, b, (short)0, c, false, false);
}

// A fragment (16x32), row-major tile, stride lda (bf16 elems).
// lane L: row=L&15; elems 0..15 -> K=16g..
__device__ __forceinline__ v16bf load_fragA(const bf16* base, int lda, int row, int g) {
    FragBF f;
    const bf16* p = base + row * lda + 16 * g;
    f.u2[0] = *(const uint2*)(p);
    f.u2[1] = *(const uint2*)(p + 4);
    f.u2[2] = *(const uint2*)(p + 8);
    f.u2[3] = *(const uint2*)(p + 12);
    return f.v;
}

// B fragment (32x16) from K-contiguous storage: 16 contiguous elems at K=16g.
__device__ __forceinline__ v16bf load_fragB(const bf16* p) {
    FragBF f;
    f.u2[0] = *(const uint2*)(p);
    f.u2[1] = *(const uint2*)(p + 4);
    f.u2[2] = *(const uint2*)(p + 8);
    f.u2[3] = *(const uint2*)(p + 12);
    return f.v;
}

__device__ __forceinline__ unsigned pk2(float a, float b) {
    union { bf16 h[2]; unsigned u; } x;
    x.h[0] = (bf16)a; x.h[1] = (bf16)b;
    return x.u;
}

// ---------------------------------------------------------------------------
// Tensor Data Mover: 2D bf16 tile -> LDS (6-arg builtin on this toolchain:
//   (uint32x4 g0, int32x8 g1, int32x4 g2, int32x4 g3, int32x8, i32 cpol))
// ---------------------------------------------------------------------------
#if __has_builtin(__builtin_amdgcn_tensor_load_to_lds)
#define HAVE_TDM 1
typedef unsigned v4u __attribute__((ext_vector_type(4)));
typedef int      v8i __attribute__((ext_vector_type(8)));
typedef int      v4i __attribute__((ext_vector_type(4)));

// pad_int_code: table code for dwords-before-pad; pad_amt_code: pad dwords - 1
__device__ __forceinline__ void tdm_load_2d_bf16(
    const void* gaddr, unsigned lds_off,
    unsigned tile_d0, unsigned tile_d1,
    unsigned tensor_d0, unsigned tensor_d1,
    unsigned long long stride0,
    unsigned pad_int_code, unsigned pad_amt_code)
{
    unsigned long long ga = (unsigned long long)gaddr;
    v4u g0;
    g0[0] = 1u;                                              // count=1, user
    g0[1] = lds_off;                                         // lds_addr bytes
    g0[2] = (unsigned)ga;                                    // global[31:0]
    g0[3] = (unsigned)((ga >> 32) & 0x01FFFFFFu) | (2u << 30); // type=2
    v8i g1;
    unsigned w0 = (1u << 16)                                 // data_size=2B
                | (1u << 20)                                 // pad_enable
                | (pad_int_code << 22) | (pad_amt_code << 25);
    g1[0] = (int)w0;
    g1[1] = (int)((tensor_d0 & 0xFFFFu) << 16);              // dim0[15:0]
    g1[2] = (int)((tensor_d0 >> 16) | ((tensor_d1 & 0xFFFFu) << 16));
    g1[3] = (int)((tensor_d1 >> 16) | (tile_d0 << 16));
    g1[4] = (int)(tile_d1 & 0xFFFFu);                        // tile_dim2 = 0
    g1[5] = (int)(unsigned)(stride0 & 0xFFFFFFFFull);
    g1[6] = (int)(unsigned)((stride0 >> 32) & 0xFFFFull);    // stride1 = 0
    g1[7] = 0;
    v4i z4 = {0, 0, 0, 0};
    v8i z8 = {0, 0, 0, 0, 0, 0, 0, 0};
    __builtin_amdgcn_tensor_load_to_lds(g0, g1, z4, z4, z8, 0);
}
__device__ __forceinline__ void tdm_wait() {
    __builtin_amdgcn_s_wait_tensorcnt((short)0);
}
__device__ __forceinline__ unsigned lds_off_u32(const void* p) {
    return (unsigned)(unsigned long long)p;                  // low 32 = LDS offset
}
#endif

#define BM 128
#define BN 128
#define BK 32
#define LDA 40
#define LDB 40

// ---------------------------------------------------------------------------
// Kernel 1: q/k = relu(X@W) -> bf16 [B,H,S,64]; v -> transposed [B,H,64,S].
// q scaled by 1/8 (folds 1/sqrt(dk) into scores).
// ---------------------------------------------------------------------------
__global__ __launch_bounds__(256)
void qkv_proj_kernel(const float* __restrict__ Xq, const float* __restrict__ Xk,
                     const float* __restrict__ Xv, const float* __restrict__ Wq,
                     const float* __restrict__ Wk, const float* __restrict__ Wv,
                     bf16* __restrict__ oq, bf16* __restrict__ ok, bf16* __restrict__ ovT)
{
    __shared__ __align__(16) bf16 sA[BM * LDA];
    __shared__ __align__(16) bf16 sBt[BN * LDB];

    const int z = blockIdx.z;
    const float* X = (z == 0) ? Xq : (z == 1) ? Xk : Xv;
    const float* W = (z == 0) ? Wq : (z == 1) ? Wk : Wv;

    const int tid  = threadIdx.x;
    const int lane = tid & 31;
    const int w    = tid >> 5;     // 0..7
    const int wm   = w >> 1;       // 0..3  (32-row strip)
    const int wn   = w & 1;        // 0..1  (64-col strip)
    const int g    = lane >> 4;
    const int c    = lane & 15;
    const int blockM = blockIdx.y * BM;
    const int blockN = blockIdx.x * BN;

    v8f acc[2][4] = {};

    for (int k0 = 0; k0 < 1024; k0 += BK) {
        __syncthreads();
        // A tile: 128x32 f32 -> bf16, packed b64 stores
        #pragma unroll
        for (int i = 0; i < 4; ++i) {
            int idx = tid + 256 * i;
            int row = idx >> 3;
            int col = (idx & 7) << 2;
            const float4 f = *(const float4*)(X + (size_t)(blockM + row) * 1024 + k0 + col);
            *(uint2*)&sA[row * LDA + col] = make_uint2(pk2(f.x, f.y), pk2(f.z, f.w));
        }
        // B tile transposed: 32(K) x 128(N): each thread owns a 4x4 block
        {
            int kq = (tid >> 5) << 2;     // 0..28
            int n  = (tid & 31) << 2;     // 0..124
            const float* Wp = W + (size_t)(k0 + kq) * 1024 + blockN + n;
            const float4 r0 = *(const float4*)(Wp);
            const float4 r1 = *(const float4*)(Wp + 1024);
            const float4 r2 = *(const float4*)(Wp + 2048);
            const float4 r3 = *(const float4*)(Wp + 3072);
            *(uint2*)&sBt[(n + 0) * LDB + kq] = make_uint2(pk2(r0.x, r1.x), pk2(r2.x, r3.x));
            *(uint2*)&sBt[(n + 1) * LDB + kq] = make_uint2(pk2(r0.y, r1.y), pk2(r2.y, r3.y));
            *(uint2*)&sBt[(n + 2) * LDB + kq] = make_uint2(pk2(r0.z, r1.z), pk2(r2.z, r3.z));
            *(uint2*)&sBt[(n + 3) * LDB + kq] = make_uint2(pk2(r0.w, r1.w), pk2(r2.w, r3.w));
        }
        __syncthreads();

        v16bf fb[4];
        #pragma unroll
        for (int ni = 0; ni < 4; ++ni)
            fb[ni] = load_fragB(&sBt[(wn * 64 + ni * 16 + c) * LDB + 16 * g]);
        #pragma unroll
        for (int mi = 0; mi < 2; ++mi) {
            v16bf fa = load_fragA(sA, LDA, wm * 32 + mi * 16 + c, g);
            #pragma unroll
            for (int ni = 0; ni < 4; ++ni)
                acc[mi][ni] = wmma_bf16(fa, fb[ni], acc[mi][ni]);
        }
    }

    const float qscale = (z == 0) ? 0.125f : 1.0f;
    #pragma unroll
    for (int mi = 0; mi < 2; ++mi) {
        #pragma unroll
        for (int ni = 0; ni < 4; ++ni) {
            int n = blockN + wn * 64 + ni * 16 + c;
            int h = n >> 6, d = n & 63;
            #pragma unroll
            for (int j = 0; j < 8; ++j) {
                int row = blockM + wm * 32 + mi * 16 + j + 8 * g;
                int b = row >> 11, s = row & 2047;
                float v = acc[mi][ni][j];
                v = v > 0.0f ? v : 0.0f;
                if (z == 2) {
                    // V transposed: [ (b*16+h)*64 + d ][ s ]
                    ovT[((size_t)((b * 16 + h) * 64 + d)) * 2048 + s] = (bf16)v;
                } else {
                    bf16* out = (z == 0) ? oq : ok;
                    out[(((size_t)b * 16 + h) * 2048 + s) * 64 + d] = (bf16)(v * qscale);
                }
            }
        }
    }
}

// ---------------------------------------------------------------------------
// Kernel 2: flash attention. Workgroup = one (b,h) x 256 queries (32/wave).
// 64-key tiles staged to LDS (TDM when available), 32-key WMMA sub-blocks.
// ---------------------------------------------------------------------------
#define AKB 64
#define LDK 72
#define LDV 72
#define LDP 40

__global__ __launch_bounds__(256)
void attention_kernel(const bf16* __restrict__ qb, const bf16* __restrict__ kbuf,
                      const bf16* __restrict__ vtb, bf16* __restrict__ ctx)
{
    __shared__ __align__(16) bf16 sK [64 * LDK];   // [key][d]
    __shared__ __align__(16) bf16 sVt[64 * LDV];   // [d][key]
    __shared__ __align__(16) bf16 sP [8 * 32 * LDP];

    const int tid  = threadIdx.x;
    const int lane = tid & 31;
    const int w    = tid >> 5;
    const int g    = lane >> 4;
    const int c    = lane & 15;
    const int bh   = blockIdx.y;
    const int b    = bh >> 4, h = bh & 15;
    const int qtile = blockIdx.x * 256 + w * 32;

    const bf16* qp  = qb   + (size_t)bh * 2048 * 64;
    const bf16* kp  = kbuf + (size_t)bh * 2048 * 64;
    const bf16* vtp = vtb  + (size_t)bh * 64 * 2048;

    // Q fragments (2 row-tiles x 2 K-steps), loaded once from global.
    FragBF fq[2][2];
    #pragma unroll
    for (int mi = 0; mi < 2; ++mi) {
        int r = qtile + mi * 16 + c;
        #pragma unroll
        for (int ks = 0; ks < 2; ++ks) {
            const bf16* p = qp + (size_t)r * 64 + 32 * ks + 16 * g;
            fq[mi][ks].u2[0] = *(const uint2*)(p);
            fq[mi][ks].u2[1] = *(const uint2*)(p + 4);
            fq[mi][ks].u2[2] = *(const uint2*)(p + 8);
            fq[mi][ks].u2[3] = *(const uint2*)(p + 12);
        }
    }

    v8f acc[2][4] = {};
    float m[2][8], l[2][8];
    #pragma unroll
    for (int mi = 0; mi < 2; ++mi)
        for (int j = 0; j < 8; ++j) { m[mi][j] = -1e30f; l[mi][j] = 0.0f; }

    bf16* pw = &sP[w * 32 * LDP];

    for (int k0 = 0; k0 < 2048; k0 += AKB) {
        __syncthreads();
#ifdef HAVE_TDM
        if (w == 0) {
            // K tile: 64 keys x 64 d (row-major, stride 64); pad 32dw -> +4dw
            tdm_load_2d_bf16(kp + (size_t)k0 * 64, lds_off_u32(sK),
                             64, AKB, 64, 2048, 64, 4, 3);
            // Vt tile: 64 d rows x 64 keys from [d][2048]
            tdm_load_2d_bf16(vtp + k0, lds_off_u32(sVt),
                             64, 64, 2048, 64, 2048, 4, 3);
            tdm_wait();
        }
#else
        #pragma unroll
        for (int i = 0; i < 2; ++i) {
            int idx = tid + 256 * i;
            int row = idx >> 3, seg = (idx & 7) << 3;
            *(uint4*)&sK[row * LDK + seg] =
                *(const uint4*)(kp + (size_t)(k0 + row) * 64 + seg);
        }
        #pragma unroll
        for (int i = 0; i < 2; ++i) {
            int idx = tid + 256 * i;
            int row = idx >> 3, seg = (idx & 7) << 3;
            *(uint4*)&sVt[row * LDV + seg] =
                *(const uint4*)(vtp + (size_t)row * 2048 + k0 + seg);
        }
#endif
        __syncthreads();

        #pragma unroll
        for (int kk = 0; kk < AKB; kk += 32) {
            // scores: Q(32x64) @ K^T(64x32)
            v8f s[2][2] = {};
            #pragma unroll
            for (int ks = 0; ks < 2; ++ks) {
                #pragma unroll
                for (int ni = 0; ni < 2; ++ni) {
                    v16bf fk = load_fragB(&sK[(kk + ni * 16 + c) * LDK + 32 * ks + 16 * g]);
                    s[0][ni] = wmma_bf16(fq[0][ks].v, fk, s[0][ni]);
                    s[1][ni] = wmma_bf16(fq[1][ks].v, fk, s[1][ni]);
                }
            }
            // online softmax per row-tile
            #pragma unroll
            for (int mi = 0; mi < 2; ++mi) {
                float alpha[8];
                #pragma unroll
                for (int j = 0; j < 8; ++j) {
                    float mx = fmaxf(m[mi][j], fmaxf(s[mi][0][j], s[mi][1][j]));
                    #pragma unroll
                    for (int off = 1; off < 16; off <<= 1)
                        mx = fmaxf(mx, __shfl_xor(mx, off, 32));
                    float p0 = __expf(s[mi][0][j] - mx);
                    float p1 = __expf(s[mi][1][j] - mx);
                    float rs = p0 + p1;
                    #pragma unroll
                    for (int off = 1; off < 16; off <<= 1)
                        rs += __shfl_xor(rs, off, 32);
                    float a = __expf(m[mi][j] - mx);
                    l[mi][j] = l[mi][j] * a + rs;
                    m[mi][j] = mx;
                    alpha[j] = a;
                    pw[(mi * 16 + j + 8 * g) * LDP + c]      = (bf16)p0;
                    pw[(mi * 16 + j + 8 * g) * LDP + c + 16] = (bf16)p1;
                }
                #pragma unroll
                for (int di = 0; di < 4; ++di)
                    #pragma unroll
                    for (int j = 0; j < 8; ++j)
                        acc[mi][di][j] = acc[mi][di][j] * alpha[j];
            }
            // P(32x32) @ V(32x64); per-wave LDS P (same-wave LDS is in-order)
            v16bf fp0 = load_fragA(pw, LDP, c, g);
            v16bf fp1 = load_fragA(pw, LDP, 16 + c, g);
            #pragma unroll
            for (int di = 0; di < 4; ++di) {
                v16bf fv = load_fragB(&sVt[(di * 16 + c) * LDV + kk + 16 * g]);
                acc[0][di] = wmma_bf16(fp0, fv, acc[0][di]);
                acc[1][di] = wmma_bf16(fp1, fv, acc[1][di]);
            }
        }
    }

    // finalize: ctx/l -> [B, S, H*64] bf16
    #pragma unroll
    for (int mi = 0; mi < 2; ++mi)
        #pragma unroll
        for (int di = 0; di < 4; ++di)
            #pragma unroll
            for (int j = 0; j < 8; ++j) {
                int qrow = qtile + mi * 16 + j + 8 * g;
                float v = acc[mi][di][j] / l[mi][j];
                ctx[((size_t)b * 2048 + qrow) * 1024 + h * 64 + di * 16 + c] = (bf16)v;
            }
}

// ---------------------------------------------------------------------------
// Kernel 3: out = relu(ctx_bf16 @ Wo) -> f32. A tile staged via TDM.
// ---------------------------------------------------------------------------
__global__ __launch_bounds__(256)
void out_proj_kernel(const bf16* __restrict__ Xc, const float* __restrict__ Wo,
                     float* __restrict__ out)
{
    __shared__ __align__(16) bf16 sA[BM * LDA];
    __shared__ __align__(16) bf16 sBt[BN * LDB];

    const int tid  = threadIdx.x;
    const int lane = tid & 31;
    const int w    = tid >> 5;
    const int wm   = w >> 1;
    const int wn   = w & 1;
    const int g    = lane >> 4;
    const int c    = lane & 15;
    const int blockM = blockIdx.y * BM;
    const int blockN = blockIdx.x * BN;

    v8f acc[2][4] = {};

    for (int k0 = 0; k0 < 1024; k0 += BK) {
        __syncthreads();
#ifdef HAVE_TDM
        if (w == 0) {
            // A tile: 128 rows x 32 bf16 cols, global stride 1024.
            // pad: 16 dwords (code 3) -> +4 dwords (code 3): LDS stride 40.
            tdm_load_2d_bf16(Xc + (size_t)blockM * 1024 + k0, lds_off_u32(sA),
                             32, BM, 1024, 8192, 1024, 3, 3);
        }
#else
        #pragma unroll
        for (int i = 0; i < 2; ++i) {
            int idx = tid + 256 * i;
            int row = idx >> 2;
            int col = (idx & 3) << 3;
            *(uint4*)&sA[row * LDA + col] =
                *(const uint4*)(Xc + (size_t)(blockM + row) * 1024 + k0 + col);
        }
#endif
        // Wo tile transposed: 32x128 f32 -> bf16, 4x4 blocks
        {
            int kq = (tid >> 5) << 2;
            int n  = (tid & 31) << 2;
            const float* Wp = Wo + (size_t)(k0 + kq) * 1024 + blockN + n;
            const float4 r0 = *(const float4*)(Wp);
            const float4 r1 = *(const float4*)(Wp + 1024);
            const float4 r2 = *(const float4*)(Wp + 2048);
            const float4 r3 = *(const float4*)(Wp + 3072);
            *(uint2*)&sBt[(n + 0) * LDB + kq] = make_uint2(pk2(r0.x, r1.x), pk2(r2.x, r3.x));
            *(uint2*)&sBt[(n + 1) * LDB + kq] = make_uint2(pk2(r0.y, r1.y), pk2(r2.y, r3.y));
            *(uint2*)&sBt[(n + 2) * LDB + kq] = make_uint2(pk2(r0.z, r1.z), pk2(r2.z, r3.z));
            *(uint2*)&sBt[(n + 3) * LDB + kq] = make_uint2(pk2(r0.w, r1.w), pk2(r2.w, r3.w));
        }
#ifdef HAVE_TDM
        if (w == 0) __builtin_amdgcn_s_wait_tensorcnt((short)0);
#endif
        __syncthreads();

        v16bf fb[4];
        #pragma unroll
        for (int ni = 0; ni < 4; ++ni)
            fb[ni] = load_fragB(&sBt[(wn * 64 + ni * 16 + c) * LDB + 16 * g]);
        #pragma unroll
        for (int mi = 0; mi < 2; ++mi) {
            v16bf fa = load_fragA(sA, LDA, wm * 32 + mi * 16 + c, g);
            #pragma unroll
            for (int ni = 0; ni < 4; ++ni)
                acc[mi][ni] = wmma_bf16(fa, fb[ni], acc[mi][ni]);
        }
    }

    #pragma unroll
    for (int mi = 0; mi < 2; ++mi)
        #pragma unroll
        for (int ni = 0; ni < 4; ++ni) {
            int n = blockN + wn * 64 + ni * 16 + c;
            #pragma unroll
            for (int j = 0; j < 8; ++j) {
                int row = blockM + wm * 32 + mi * 16 + j + 8 * g;
                float v = acc[mi][ni][j];
                out[(size_t)row * 1024 + n] = v > 0.0f ? v : 0.0f;
            }
        }
}

// ---------------------------------------------------------------------------
extern "C" void kernel_launch(void* const* d_in, const int* in_sizes, int n_in,
                              void* d_out, int out_size, void* d_ws, size_t ws_size,
                              hipStream_t stream) {
    const float* Xq = (const float*)d_in[0];
    const float* Xk = (const float*)d_in[1];
    const float* Xv = (const float*)d_in[2];
    const float* Wq = (const float*)d_in[3];
    const float* Wk = (const float*)d_in[4];
    const float* Wv = (const float*)d_in[5];
    const float* Wo = (const float*)d_in[6];

    const size_t NELEM = (size_t)8192 * 1024;
    bf16* qw  = (bf16*)d_ws;
    bf16* kw  = qw + NELEM;
    bf16* vtw = kw + NELEM;   // transposed V: [B,H,64,S]
    bf16* cw  = vtw + NELEM;

    dim3 g1(1024 / BN, 8192 / BM, 3);
    qkv_proj_kernel<<<g1, 256, 0, stream>>>(Xq, Xk, Xv, Wq, Wk, Wv, qw, kw, vtw);

    dim3 g2(2048 / 256, 64);
    attention_kernel<<<g2, 256, 0, stream>>>(qw, kw, vtw, cw);

    dim3 g3(1024 / BN, 8192 / BM);
    out_proj_kernel<<<g3, 256, 0, stream>>>(cw, Wo, (float*)d_out);
}
